// CrossModalAttention_83872121356769
// MI455X (gfx1250) — matrix operand order repaired and checked by reading, not compiled, not run on hardware
//
#include <hip/hip_runtime.h>
#include <hip/hip_bf16.h>

typedef __attribute__((ext_vector_type(16))) _Float16 v16h;
typedef __attribute__((ext_vector_type(4)))  _Float16 v4h;
typedef __attribute__((ext_vector_type(4)))  float    v4f;
typedef __attribute__((ext_vector_type(8)))  float    v8f;

#define D_MODEL 1024
#define N_HEADS 16
#define D_K     64
#define BATCH   4
#define SEQ     1024
#define NROWS   (BATCH * SEQ)          // 4096 rows for all row-major GEMMs
#define TS16    (16 * D_MODEL)         // B-tile row-block stride (halves)

static __device__ __forceinline__ v8f wmma_f16(v16h a, v16h b, v8f c) {
    return __builtin_amdgcn_wmma_f32_16x16x32_f16(
        /*neg_a=*/false, a, /*neg_b=*/false, b,
        /*c_mod=*/(short)0, c, /*reuse_a=*/false, /*reuse_b=*/false);
}

// A-operand (16x32 f16): two contiguous 8-half (16B) chunks. `p` = row + kb + 8*lhi.
static __device__ __forceinline__ v16h load_a_f16(const _Float16* __restrict__ p) {
    v16h a;
#pragma unroll
    for (int i = 0; i < 8; ++i) { a[i] = p[i]; a[8 + i] = p[16 + i]; }
    return a;
}
// B-operand (32x16): two contiguous 8-half (16B) chunks of one column. `p` = col + kb + 8*lhi.
static __device__ __forceinline__ v16h load_b_f16(const _Float16* __restrict__ p) {
    v16h b;
#pragma unroll
    for (int i = 0; i < 8; ++i) { b[i] = p[i]; b[8 + i] = p[16 + i]; }
    return b;
}

// ---------------------------------------------------------------------------
// Kernel 0: f32 -> f16 elementwise convert (vectorized 4-wide)
// ---------------------------------------------------------------------------
__global__ __launch_bounds__(256) void cvt_f32_to_f16_kernel(
    const float* __restrict__ src, _Float16* __restrict__ dst, int n4)
{
    for (int i = blockIdx.x * 256 + threadIdx.x; i < n4; i += gridDim.x * 256) {
        v4f x = ((const v4f*)src)[i];
        v4h o;
#pragma unroll
        for (int j = 0; j < 4; ++j) o[j] = (_Float16)x[j];
        ((v4h*)dst)[i] = o;
    }
}

// Shared 16x128 GEMM core with ping-pong double buffering (2-tile lookahead).
// wbase = W + (n0+llo)*D_MODEL + 8*lhi ; arow = X row base + 8*lhi.
static __device__ __forceinline__ void gemm_16x128_f16(
    const _Float16* __restrict__ arow, const _Float16* __restrict__ wbase, v8f acc[8])
{
    v16h b0 = load_b_f16(wbase);                 // even tiles
    v16h b1 = load_b_f16(wbase + TS16);          // odd tiles
    v16h aA = load_a_f16(arow);
    v16h aB = load_a_f16(arow + 32);

    auto substep = [&](int kbc, int kbn, v16h& aCur) {
#pragma unroll
        for (int t = 0; t < 8; ++t) {
            if (t & 1) {
                acc[t] = wmma_f16(aCur, b1, acc[t]);
                b1 = (t < 6) ? load_b_f16(wbase + (size_t)(t + 2) * TS16 + kbc)
                             : load_b_f16(wbase + TS16 + kbn);
            } else {
                acc[t] = wmma_f16(aCur, b0, acc[t]);
                b0 = (t < 6) ? load_b_f16(wbase + (size_t)(t + 2) * TS16 + kbc)
                             : load_b_f16(wbase + kbn);
            }
        }
    };

    for (int kb = 0; kb < D_MODEL; kb += 64) {
        const int kb2 = kb + 32;
        const int kn2 = (kb + 64 < D_MODEL) ? kb + 64 : kb;     // clamped lookahead
        const int kn3 = (kb + 96 < D_MODEL) ? kb + 96 : kb2;
        substep(kb, kb2, aA);
        aA = load_a_f16(arow + kn2);     // reload after last read; used 8 wmmas later
        substep(kb2, kn2, aB);
        aB = load_a_f16(arow + kn3);
    }
}

// ---------------------------------------------------------------------------
// Kernel 1: Q = q@Wq^T, K = k@Wk^T, V = v@Wv^T (all-f16 operands, f32 accum)
// One wave -> 16(M) x 128(N). V stored transposed per head: Vt[b][h][d][k].
// ---------------------------------------------------------------------------
__global__ __launch_bounds__(256) void qkv_gemm_kernel(
    const _Float16* __restrict__ qh, const _Float16* __restrict__ kh,
    const _Float16* __restrict__ vh,
    const _Float16* __restrict__ Wqh, const _Float16* __restrict__ Wkh,
    const _Float16* __restrict__ Wvh,
    _Float16* __restrict__ Qo, _Float16* __restrict__ Ko, _Float16* __restrict__ Vt)
{
    const int lane = threadIdx.x & 31;
    const int wv   = threadIdx.x >> 5;
    const int g    = blockIdx.x * 8 + wv;          // global wave id
    const int mat  = g >> 11;                      // / 2048
    const int rem  = g & 2047;
    const int mt   = rem >> 3;                     // 0..255
    const int ns   = rem & 7;                      // 0..7
    const _Float16* X = (mat == 0) ? qh  : (mat == 1) ? kh  : vh;
    const _Float16* W = (mat == 0) ? Wqh : (mat == 1) ? Wkh : Wvh;
    const int m0 = mt * 16, n0 = ns * 128;
    const int lhi = lane >> 4, llo = lane & 15;

    v8f acc[8] = {};
    const _Float16* arow  = X + (size_t)(m0 + llo) * D_MODEL + 8 * lhi;
    const _Float16* wbase = W + (size_t)(n0 + llo) * D_MODEL + 8 * lhi;
    gemm_16x128_f16(arow, wbase, acc);

    if (mat < 2) {
        _Float16* O = (mat == 0) ? Qo : Ko;
#pragma unroll
        for (int t = 0; t < 8; ++t)
#pragma unroll
            for (int vv = 0; vv < 8; ++vv)
                O[(size_t)(m0 + vv + 8 * lhi) * D_MODEL + n0 + t * 16 + llo] =
                    (_Float16)acc[t][vv];
    } else {
        // V transposed per head: Vt[((b*16+h)*64 + d)*1024 + kseq]
#pragma unroll
        for (int t = 0; t < 8; ++t) {
            const int n = n0 + t * 16 + llo;
            const int h = n >> 6, d = n & 63;
#pragma unroll
            for (int vv = 0; vv < 8; ++vv) {
                const int m = m0 + vv + 8 * lhi;
                Vt[((size_t)((m >> 10) * N_HEADS + h) * D_K + d) * SEQ + (m & 1023)] =
                    (_Float16)acc[t][vv];
            }
        }
    }
}

// ---------------------------------------------------------------------------
// Kernel 2: per (b, h, 16-row q tile): S = QK^T*scale (masked), softmax,
// attn -> d_out (f32) and LDS (f16), ctx = attn@V -> ws (f16).
// Block = 8 waves; wave w owns score columns [w*128, w*128+128).
// ---------------------------------------------------------------------------
__global__ __launch_bounds__(256) void attn_softmax_av_kernel(
    const _Float16* __restrict__ Qo, const _Float16* __restrict__ Ko,
    const _Float16* __restrict__ Vt, const int* __restrict__ mask,
    const float* __restrict__ cmw, float* __restrict__ attn_out,
    _Float16* __restrict__ ctx)
{
    __shared__ _Float16 attn_lds[16][1032];   // padded row stride (16B aligned)
    __shared__ float    ctx_lds[16][64];
    __shared__ float    red_max[16][8];
    __shared__ float    red_sum[16][8];

    const int lane = threadIdx.x & 31;
    const int wv   = threadIdx.x >> 5;
    const int b    = blockIdx.x >> 10;        // / (16*64)
    const int h    = (blockIdx.x >> 6) & 15;
    const int qt   = blockIdx.x & 63;
    const int q0   = qt * 16;
    const int lhi  = lane >> 4, llo = lane & 15;
    const int kc0  = wv * 128;

    // zero ctx accumulation buffer
    for (int i = threadIdx.x; i < 16 * 64; i += 256) ((float*)ctx_lds)[i] = 0.0f;

    // A operand: 16 q-rows x 64 d, two 16x32 slabs (reused across 8 B tiles)
    const _Float16* qrow = Qo + ((size_t)(b * SEQ + q0 + llo)) * D_MODEL + h * D_K + 8 * lhi;
    v16h a0 = load_a_f16(qrow);
    v16h a1 = load_a_f16(qrow + 32);

    // QK^T: 16 B-slabs (8 tiles x lo/hi), ping-pong with 2-slab lookahead.
    // slab j: tile t=j>>1, half (j&1): ptr = kbase + (j>>1)*TS16 + (j&1)*32
    const _Float16* kbase = Ko + ((size_t)(b * SEQ + kc0 + llo)) * D_MODEL
                             + h * D_K + 8 * lhi;
    v8f acc[8] = {};
    {
        v16h bE = load_b_f16(kbase);          // j=0
        v16h bO = load_b_f16(kbase + 32);     // j=1
#pragma unroll
        for (int j = 0; j < 16; ++j) {
            const int jn = (j + 2 < 16) ? j + 2 : j;   // clamped (redundant reload)
            const _Float16* np = kbase + (size_t)(jn >> 1) * TS16 + (jn & 1) * 32;
            if (j & 1) { acc[j >> 1] = wmma_f16(a1, bO, acc[j >> 1]); bO = load_b_f16(np); }
            else       { acc[j >> 1] = wmma_f16(a0, bE, acc[j >> 1]); bE = load_b_f16(np); }
        }
    }

    // scale + per-head weight + mask
    const float scale = 0.125f * cmw[h];      // 1/sqrt(64) * cmw
    const int*  mrow  = mask + (size_t)(b * SEQ + q0) * SEQ;
#pragma unroll
    for (int t = 0; t < 8; ++t) {
        const int cc = kc0 + t * 16 + llo;
#pragma unroll
        for (int vv = 0; vv < 8; ++vv) {
            const int qq = vv + 8 * lhi;
            float s = acc[t][vv] * scale;
            if (mrow[(size_t)qq * SEQ + cc] == 0) s = -1e9f;
            acc[t][vv] = s;
        }
    }

    // row max: per-lane over 8 tiles, then 16-lane butterfly, then cross-wave LDS
    float rmx[8];
#pragma unroll
    for (int vv = 0; vv < 8; ++vv) {
        float m = acc[0][vv];
#pragma unroll
        for (int t = 1; t < 8; ++t) m = fmaxf(m, acc[t][vv]);
#pragma unroll
        for (int off = 1; off < 16; off <<= 1) m = fmaxf(m, __shfl_xor(m, off, 32));
        rmx[vv] = m;
    }
    if (llo == 0)
#pragma unroll
        for (int vv = 0; vv < 8; ++vv) red_max[vv + 8 * lhi][wv] = rmx[vv];
    __syncthreads();

    float gmx[8];
#pragma unroll
    for (int vv = 0; vv < 8; ++vv) {
        const int row = vv + 8 * lhi;
        float m = red_max[row][0];
#pragma unroll
        for (int w = 1; w < 8; ++w) m = fmaxf(m, red_max[row][w]);
        gmx[vv] = m;
    }

    // exp + row sum
    float rsum[8] = {};
#pragma unroll
    for (int t = 0; t < 8; ++t)
#pragma unroll
        for (int vv = 0; vv < 8; ++vv) {
            float e = __expf(acc[t][vv] - gmx[vv]);
            acc[t][vv] = e;
            rsum[vv] += e;
        }
#pragma unroll
    for (int vv = 0; vv < 8; ++vv)
#pragma unroll
        for (int off = 1; off < 16; off <<= 1) rsum[vv] += __shfl_xor(rsum[vv], off, 32);
    if (llo == 0)
#pragma unroll
        for (int vv = 0; vv < 8; ++vv) red_sum[vv + 8 * lhi][wv] = rsum[vv];
    __syncthreads();

    float inv[8];
#pragma unroll
    for (int vv = 0; vv < 8; ++vv) {
        const int row = vv + 8 * lhi;
        float s = red_sum[row][0];
#pragma unroll
        for (int w = 1; w < 8; ++w) s += red_sum[row][w];
        inv[vv] = 1.0f / s;
    }

    // write attn (f32 to d_out, f16 to LDS in row-major for A-operand reuse)
    float* aout = attn_out + ((size_t)(b * N_HEADS + h) * SEQ + q0) * SEQ;
#pragma unroll
    for (int t = 0; t < 8; ++t) {
        const int cc = kc0 + t * 16 + llo;
#pragma unroll
        for (int vv = 0; vv < 8; ++vv) {
            const int qq = vv + 8 * lhi;
            const float p = acc[t][vv] * inv[vv];
            aout[(size_t)qq * SEQ + cc] = p;
            attn_lds[qq][cc] = (_Float16)p;
        }
    }
    __syncthreads();

    // ctx partial over k in [kc0, kc0+128): 16 tiles (kk,t), ping-pong B.
    // tile j: kk=j>>2, t=j&3: B ptr = vbase + (t*16+llo)*SEQ + kc0 + kk*32 + 8*lhi
    v8f cacc[4] = {};
    const _Float16* vbase = Vt + (size_t)(b * N_HEADS + h) * D_K * SEQ
                             + (size_t)llo * SEQ + kc0 + 8 * lhi;
    {
        v16h bE = load_b_f16(vbase);                       // j=0 (kk0,t0)
        v16h bO = load_b_f16(vbase + (size_t)16 * SEQ);    // j=1 (kk0,t1)
        v16h a  = load_a_f16(&attn_lds[llo][kc0 + 8 * lhi]);
#pragma unroll
        for (int j = 0; j < 16; ++j) {
            const int kk = j >> 2, t = j & 3;
            if (t == 0 && kk > 0) a = load_a_f16(&attn_lds[llo][kc0 + kk * 32 + 8 * lhi]);
            const int jn = (j + 2 < 16) ? j + 2 : j;       // clamped
            const _Float16* np = vbase + (size_t)(jn & 3) * 16 * SEQ + (jn >> 2) * 32;
            if (j & 1) { cacc[t] = wmma_f16(a, bO, cacc[t]); bO = load_b_f16(np); }
            else       { cacc[t] = wmma_f16(a, bE, cacc[t]); bE = load_b_f16(np); }
        }
    }
#pragma unroll
    for (int t = 0; t < 4; ++t)
#pragma unroll
        for (int vv = 0; vv < 8; ++vv)
            atomicAdd(&ctx_lds[vv + 8 * lhi][t * 16 + llo], cacc[t][vv]);
    __syncthreads();

    for (int i = threadIdx.x; i < 16 * 64; i += 256) {
        const int r = i >> 6, c = i & 63;
        ctx[((size_t)(b * SEQ + q0 + r)) * D_MODEL + h * D_K + c] = (_Float16)ctx_lds[r][c];
    }
}

// ---------------------------------------------------------------------------
// Kernel 3: y = LayerNorm(ctx @ Wo^T + bo + query). One block = 16 rows;
// wave w owns columns [w*128, w*128+128). LN via shuffle + tiny LDS reduce.
// ---------------------------------------------------------------------------
__global__ __launch_bounds__(256) void oproj_ln_kernel(
    const _Float16* __restrict__ ctx, const _Float16* __restrict__ Woh,
    const float* __restrict__ bo, const float* __restrict__ query,
    const float* __restrict__ gamma, const float* __restrict__ beta,
    float* __restrict__ y)
{
    __shared__ float red1[16][8];
    __shared__ float red2[16][8];

    const int lane = threadIdx.x & 31;
    const int wv   = threadIdx.x >> 5;
    const int m0   = blockIdx.x * 16;
    const int lhi  = lane >> 4, llo = lane & 15;
    const int n0   = wv * 128;

    v8f acc[8] = {};
    const _Float16* arow  = ctx + (size_t)(m0 + llo) * D_MODEL + 8 * lhi;
    const _Float16* wbase = Woh + (size_t)(n0 + llo) * D_MODEL + 8 * lhi;
    gemm_16x128_f16(arow, wbase, acc);

    // x = out + bias + residual; accumulate row sums / sumsq
    float s1[8] = {}, s2[8] = {};
#pragma unroll
    for (int t = 0; t < 8; ++t) {
        const int n = n0 + t * 16 + llo;
        const float bias = bo[n];
#pragma unroll
        for (int vv = 0; vv < 8; ++vv) {
            const int m = m0 + vv + 8 * lhi;
            float x = acc[t][vv] + bias + query[(size_t)m * D_MODEL + n];
            acc[t][vv] = x;
            s1[vv] += x;
            s2[vv] += x * x;
        }
    }
#pragma unroll
    for (int vv = 0; vv < 8; ++vv)
#pragma unroll
        for (int off = 1; off < 16; off <<= 1) {
            s1[vv] += __shfl_xor(s1[vv], off, 32);
            s2[vv] += __shfl_xor(s2[vv], off, 32);
        }
    if (llo == 0)
#pragma unroll
        for (int vv = 0; vv < 8; ++vv) {
            red1[vv + 8 * lhi][wv] = s1[vv];
            red2[vv + 8 * lhi][wv] = s2[vv];
        }
    __syncthreads();

    float mu[8], rs[8];
#pragma unroll
    for (int vv = 0; vv < 8; ++vv) {
        const int row = vv + 8 * lhi;
        float t1 = 0.0f, t2 = 0.0f;
#pragma unroll
        for (int w = 0; w < 8; ++w) { t1 += red1[row][w]; t2 += red2[row][w]; }
        const float m = t1 * (1.0f / D_MODEL);
        const float var = t2 * (1.0f / D_MODEL) - m * m;
        mu[vv] = m;
        rs[vv] = rsqrtf(var + 1e-5f);
    }
#pragma unroll
    for (int t = 0; t < 8; ++t) {
        const int n = n0 + t * 16 + llo;
        const float g = gamma[n], be = beta[n];
#pragma unroll
        for (int vv = 0; vv < 8; ++vv) {
            const int m = m0 + vv + 8 * lhi;
            y[(size_t)m * D_MODEL + n] = (acc[t][vv] - mu[vv]) * rs[vv] * g + be;
        }
    }
}

// ---------------------------------------------------------------------------
extern "C" void kernel_launch(void* const* d_in, const int* in_sizes, int n_in,
                              void* d_out, int out_size, void* d_ws, size_t ws_size,
                              hipStream_t stream) {
    const float* query = (const float*)d_in[0];
    const float* key   = (const float*)d_in[1];
    const float* value = (const float*)d_in[2];
    const int*   mask  = (const int*)d_in[3];
    const float* Wq    = (const float*)d_in[4];
    const float* Wk    = (const float*)d_in[5];
    const float* Wv    = (const float*)d_in[6];
    const float* Wo    = (const float*)d_in[7];
    const float* bo    = (const float*)d_in[8];
    const float* cmw   = (const float*)d_in[9];
    const float* gamma = (const float*)d_in[10];
    const float* beta  = (const float*)d_in[11];

    float* y    = (float*)d_out;                               // [4,1024,1024]
    float* attn = y + (size_t)BATCH * SEQ * D_MODEL;           // [4,16,1024,1024]

    const size_t ACT = (size_t)NROWS * D_MODEL;                // 4M elements
    const size_t WEL = (size_t)D_MODEL * D_MODEL;              // 1M elements
    _Float16* p   = (_Float16*)d_ws;
    _Float16* qh  = p;            p += ACT;
    _Float16* kh  = p;            p += ACT;
    _Float16* vh  = p;            p += ACT;
    _Float16* Wqh = p;            p += WEL;
    _Float16* Wkh = p;            p += WEL;
    _Float16* Wvh = p;            p += WEL;
    _Float16* Woh = p;            p += WEL;
    _Float16* Qh  = p;            p += ACT;
    _Float16* Kh  = p;            p += ACT;
    _Float16* Vt  = p;            p += ACT;
    _Float16* Ch  = p;            p += ACT;   // total 32M halves = 64 MB

    // one-time f32 -> f16 conversions (removes all cvts from GEMM hot loops)
    cvt_f32_to_f16_kernel<<<2048, 256, 0, stream>>>(query, qh, (int)(ACT / 4));
    cvt_f32_to_f16_kernel<<<2048, 256, 0, stream>>>(key,   kh, (int)(ACT / 4));
    cvt_f32_to_f16_kernel<<<2048, 256, 0, stream>>>(value, vh, (int)(ACT / 4));
    cvt_f32_to_f16_kernel<<<1024, 256, 0, stream>>>(Wq, Wqh, (int)(WEL / 4));
    cvt_f32_to_f16_kernel<<<1024, 256, 0, stream>>>(Wk, Wkh, (int)(WEL / 4));
    cvt_f32_to_f16_kernel<<<1024, 256, 0, stream>>>(Wv, Wvh, (int)(WEL / 4));
    cvt_f32_to_f16_kernel<<<1024, 256, 0, stream>>>(Wo, Woh, (int)(WEL / 4));

    // 3 mats * 256 mtiles * 8 nstrips waves / 8 waves per block = 768 blocks
    qkv_gemm_kernel<<<768, 256, 0, stream>>>(qh, kh, vh, Wqh, Wkh, Wvh, Qh, Kh, Vt);
    // B * H * (SQ/16) = 4096 blocks
    attn_softmax_av_kernel<<<4096, 256, 0, stream>>>(Qh, Kh, Vt, mask, cmw, attn, Ch);
    // NROWS/16 = 256 blocks
    oproj_ln_kernel<<<256, 256, 0, stream>>>(Ch, Woh, bo, query, gamma, beta, y);
}
